// GAT_AE_79834852098595
// MI455X (gfx1250) — compile-verified, matrix operand order checked by reading, not run on hardware
//
#include <hip/hip_runtime.h>
#include <math.h>

typedef unsigned int u32;
typedef unsigned short u16;
typedef __attribute__((ext_vector_type(8)))  float  v8f;
typedef __attribute__((ext_vector_type(16))) __bf16 v16bf;
typedef __attribute__((ext_vector_type(4)))  int    v4i;

#define NN    65536      // nodes per graph (BATCH*SEQ)
#define BB    2048       // batch
#define SEQL  32
#define HIDD  128
#define NED   131072     // edges per graph (without self loops)
#define NET   (NED + NN) // with self loops
#define LTOT  64         // 2*SEQ encoder rows

#define AS1 __attribute__((address_space(1)))
#define AS3 __attribute__((address_space(3)))

// ---------------- helpers ----------------
__device__ __forceinline__ u16 f2bf(float f) {
  u32 x = __float_as_uint(f);
  u32 r = (x + 0x7fffu + ((x >> 16) & 1u)) >> 16;   // RNE
  return (u16)r;
}
__device__ __forceinline__ u32 fenc_(float f) {     // order-preserving float->uint
  u32 b = __float_as_uint(f);
  return (b & 0x80000000u) ? ~b : (b | 0x80000000u);
}
__device__ __forceinline__ float fdec_(u32 e) {
  u32 b = (e & 0x80000000u) ? (e & 0x7fffffffu) : ~e;
  return __uint_as_float(b);
}

// ---------------- WMMA GEMM: C[M,N](+bias) = A[M,K]bf16 * Bt[N,K]bf16 ----------------
// block = 4 waves; each wave computes a 32x64 tile (2 A-frags x 4 B-frags = 8 WMMA/K-step).
// B panel [64 x K] is staged once in LDS via gfx1250 async-to-LDS copies and shared by
// all 4 waves. Rows padded by 8 elems so b128 LDS reads are bank-conflict free.
// M must be a multiple of 128, K a multiple of 32; N guarded on store.
union FragU { v16bf v; uint4 u[2]; };
union AccU  { v8f   v; float f[8]; };

__global__ __launch_bounds__(128)
void gemm_wmma(const u16* __restrict__ A, const u16* __restrict__ Bt,
               const float* __restrict__ bias, float* __restrict__ C,
               int ldc, int M, int N, int K)
{
  extern __shared__ u16 Bs[];                 // [64][K+8]
  const int KP = K + 8;                       // padded LDS row stride (elements)
  int tid  = threadIdx.x;
  int wave = tid >> 5;
  int lane = tid & 31;
  int half = lane >> 4;                       // lane half selects K sub-blocks (ISA layout)
  int l16  = lane & 15;
  int m0 = (blockIdx.y * 4 + wave) * 32;
  int n0 = blockIdx.x * 64;

  // ---- stage B panel [64 x K] -> LDS (async copy, ASYNCcnt-tracked) ----
  int cpr = K >> 3;                           // 16-byte chunks per row
  int total = 64 * cpr;                       // multiple of 256 -> no ragged tail
  for (int i = tid; i < total; i += 128) {
    int row = i / cpr, o8 = i - row * cpr;
    int rc = n0 + row;
    if (rc > N - 1) rc = N - 1;               // clamp OOB rows (cols never stored)
    const u16* gp = Bt + (size_t)rc * K + o8 * 8;
    u16* lp = Bs + (size_t)row * KP + o8 * 8;
#if __has_builtin(__builtin_amdgcn_global_load_async_to_lds_b128)
    __builtin_amdgcn_global_load_async_to_lds_b128(
        (AS1 v4i*)(reinterpret_cast<uintptr_t>(gp)),
        (AS3 v4i*)(u32)(reinterpret_cast<uintptr_t>(lp)), 0, 0);
#else
    *reinterpret_cast<uint4*>(lp) = *reinterpret_cast<const uint4*>(gp);
#endif
  }
#if __has_builtin(__builtin_amdgcn_s_wait_asynccnt)
  __builtin_amdgcn_s_wait_asynccnt(0);
#elif __has_builtin(__builtin_amdgcn_global_load_async_to_lds_b128)
  asm volatile("s_wait_asynccnt 0x0" ::: "memory");
#endif
  __syncthreads();

  AccU acc[2][4];
#pragma unroll
  for (int r = 0; r < 2; ++r)
#pragma unroll
    for (int j = 0; j < 4; ++j)
#pragma unroll
      for (int i = 0; i < 8; ++i) acc[r][j].f[i] = 0.f;

  const u16* Arow0 = A + (size_t)(m0 + l16) * K + half * 8;
  const u16* Arow1 = A + (size_t)(m0 + 16 + l16) * K + half * 8;

  for (int k0 = 0; k0 < K; k0 += 32) {
    FragU a0, a1;
    const uint4* pa0 = reinterpret_cast<const uint4*>(Arow0 + k0);
    const uint4* pa1 = reinterpret_cast<const uint4*>(Arow1 + k0);
    a0.u[0] = pa0[0];                          // K = k0+half*8 .. +7
    a0.u[1] = pa0[2];                          // K = k0+16+half*8 .. +7
    a1.u[0] = pa1[0];
    a1.u[1] = pa1[2];
    __builtin_prefetch(pa0 + 4, 0, 0);
#pragma unroll
    for (int j = 0; j < 4; ++j) {
      FragU b;
      const uint4* pb = reinterpret_cast<const uint4*>(
          Bs + (size_t)(j * 16 + l16) * KP + k0 + half * 8);
      b.u[0] = pb[0];
      b.u[1] = pb[2];
      acc[0][j].v = __builtin_amdgcn_wmma_f32_16x16x32_bf16(
          false, a0.v, false, b.v, (short)0, acc[0][j].v, false, false);
      acc[1][j].v = __builtin_amdgcn_wmma_f32_16x16x32_bf16(
          false, a1.v, false, b.v, (short)0, acc[1][j].v, false, false);
    }
  }
#pragma unroll
  for (int j = 0; j < 4; ++j) {
    int col = n0 + j * 16 + l16;
    if (col < N) {
      float bv = bias ? bias[col] : 0.f;
#pragma unroll
      for (int r = 0; r < 2; ++r)
#pragma unroll
        for (int v = 0; v < 8; ++v) {
          int row = m0 + r * 16 + half * 8 + v;  // C/D layout: M = half*8 + vgpr
          C[(size_t)row * ldc + col] = acc[r][j].f[v] + bv;
        }
    }
  }
}

// ---------------- small kernels ----------------
__global__ void fill_u32_k(u32* p, u32 v, long long n) {
  long long i = (long long)blockIdx.x * blockDim.x + threadIdx.x;
  if (i < n) p[i] = v;
}
__global__ void convbf_k(const float* __restrict__ w, u16* __restrict__ o, int n) {
  int i = blockIdx.x * blockDim.x + threadIdx.x;
  if (i < n) o[i] = f2bf(w[i]);
}
// w[K,N] row-major -> o[N,K] bf16
__global__ void transpose_bf_k(const float* __restrict__ w, u16* __restrict__ o, int K, int N) {
  int i = blockIdx.x * blockDim.x + threadIdx.x;
  if (i >= K * N) return;
  int k = i / N, n = i - k * N;
  o[(size_t)n * K + k] = f2bf(w[i]);
}
// x*sqrt(d) + positional encoding -> bf16
__global__ void prep_k(const float* __restrict__ x, u16* __restrict__ xbf, int D) {
  int i = blockIdx.x * blockDim.x + threadIdx.x;
  if (i >= NN * D) return;
  int n = i / D, k = i - n * D;
  int s = n & 31;
  float angle = (float)s * powf(10000.0f, -2.0f * (float)k / (float)D);
  float pe = (k & 1) ? cosf(angle) : sinf(angle);
  xbf[i] = f2bf(x[i] * sqrtf((float)D) + pe);
}
// a[n,h] = sum_c h[n,h,c] * att[h,c]
__global__ void attvec_k(const float* __restrict__ h, const float* __restrict__ att,
                         float* __restrict__ out, int H, int C) {
  int i = blockIdx.x * blockDim.x + threadIdx.x;
  if (i >= NN * H) return;
  int n = i / H, hd = i - n * H;
  const float* hp = h + ((size_t)n * H + hd) * C;
  const float* ap = att + hd * C;
  float s = 0.f;
  for (int c = 0; c < C; ++c) s += hp[c] * ap[c];
  out[i] = s;
}
__global__ void edge_logit_max_k(const int* __restrict__ ei,
                                 const float* __restrict__ as_, const float* __restrict__ ad_,
                                 float* __restrict__ logits, u32* __restrict__ maxb, int H) {
  int i = blockIdx.x * blockDim.x + threadIdx.x;
  if (i >= NET * H) return;
  int e = i / H, hd = i - e * H;
  int src, dst;
  if (e < NED) { src = ei[e]; dst = ei[NED + e]; } else { src = dst = e - NED; }
  float v = as_[src * H + hd] + ad_[dst * H + hd];
  v = (v > 0.f) ? v : 0.2f * v;                 // leaky_relu(0.2)
  logits[i] = v;
  atomicMax(&maxb[dst * H + hd], fenc_(v));
}
__global__ void edge_exp_k(const int* __restrict__ ei, const float* __restrict__ logits,
                           const u32* __restrict__ maxb, float* __restrict__ ebuf,
                           float* __restrict__ denom, int H) {
  int i = blockIdx.x * blockDim.x + threadIdx.x;
  if (i >= NET * H) return;
  int e = i / H, hd = i - e * H;
  int dst = (e < NED) ? ei[NED + e] : (e - NED);
  float ex = expf(logits[i] - fdec_(maxb[dst * H + hd]));
  ebuf[i] = ex;
  atomicAdd(&denom[dst * H + hd], ex);
}
// one block per edge; atomic scatter of h[src]*alpha into out[dst]
__global__ void edge_aggr_k(const int* __restrict__ ei, const float* __restrict__ h,
                            const float* __restrict__ ebuf, const float* __restrict__ denom,
                            float* __restrict__ out, int H, int C) {
  int e = blockIdx.x;
  int HC = H * C;
  int src, dst;
  if (e < NED) { src = ei[e]; dst = ei[NED + e]; } else { src = dst = e - NED; }
  for (int i = threadIdx.x; i < HC; i += blockDim.x) {
    int hd = i / C;
    float alpha = ebuf[(size_t)e * H + hd] / denom[dst * H + hd];
    atomicAdd(&out[(size_t)dst * HC + i], h[(size_t)src * HC + i] * alpha);
  }
}
__global__ void bias_elu_bf_k(const float* __restrict__ in, const float* __restrict__ bias,
                              u16* __restrict__ outbf, int W) {
  long long i = (long long)blockIdx.x * blockDim.x + threadIdx.x;
  if (i >= (long long)NN * W) return;
  int c = (int)(i % W);
  float v = in[i] + bias[c];
  v = (v > 0.f) ? v : (expf(v) - 1.f);          // ELU
  outbf[i] = f2bf(v);
}
// layer2 out + b2 -> enc[l= g*32+s, b, :] (f32 + bf16)
__global__ void enc_scatter_k(const float* __restrict__ in, const float* __restrict__ bias,
                              float* __restrict__ encf, u16* __restrict__ encb, int g) {
  long long i = (long long)blockIdx.x * blockDim.x + threadIdx.x;
  if (i >= (long long)NN * HIDD) return;
  int c = (int)(i & 127);
  int n = (int)(i >> 7);
  int b = n >> 5, s = n & 31;
  float v = in[i] + bias[c];
  size_t o = (((size_t)(g * SEQL + s) * BB) + b) * HIDD + c;
  encf[o] = v;
  encb[o] = f2bf(v);
}
__global__ void mean_k(const float* __restrict__ encf, float* __restrict__ s,
                       u16* __restrict__ sbf, int g) {
  int i = blockIdx.x * blockDim.x + threadIdx.x;
  if (i >= BB * HIDD) return;
  int b = i >> 7, c = i & 127;
  float acc = 0.f;
  for (int ss = 0; ss < SEQL; ++ss)
    acc += encf[(((size_t)(g * SEQL + ss) * BB) + b) * HIDD + c];
  acc *= (1.f / SEQL);
  s[i] = acc;
  sbf[i] = f2bf(acc);
}
// gather embedding row for each batch element; write bf16 into rin/pred slices
__global__ void emb_k(const int* __restrict__ toks, const float* __restrict__ table,
                      u16* __restrict__ rin_e, int rinW,
                      u16* __restrict__ pred_e, int predW) {
  int i = blockIdx.x * blockDim.x + threadIdx.x;
  if (i >= BB * HIDD) return;
  int b = i >> 7, c = i & 127;
  int t = toks[(size_t)b * SEQL];
  u16 v = f2bf(table[(size_t)t * HIDD + c]);
  rin_e[(size_t)b * rinW + c] = v;
  pred_e[(size_t)b * predW + c] = v;
}
// per-batch attention over L=64: softmax(q.K/sqrt(128), mask) . enc -> c (bf16 into slices)
__global__ __launch_bounds__(128)
void attn_k(const float* __restrict__ q, const float* __restrict__ Kb,
            const float* __restrict__ encf, const unsigned char* __restrict__ mask,
            u16* __restrict__ rin_c, int rinW, u16* __restrict__ pred_c, int predW) {
  int b = blockIdx.x;
  int tid = threadIdx.x;
  __shared__ float sc[LTOT];
  __shared__ float qsh[HIDD];
  qsh[tid] = q[(size_t)b * HIDD + tid];
  __syncthreads();
  if (tid < LTOT) {
    const float* kp = Kb + ((size_t)tid * BB + b) * HIDD;
    float s = 0.f;
    for (int h = 0; h < HIDD; ++h) s += qsh[h] * kp[h];
    s *= 0.088388347648318447f;                 // 1/sqrt(128)
    if (mask[b * SEQL + (tid & 31)]) s = -INFINITY;
    sc[tid] = s;
  }
  __syncthreads();
  if (tid == 0) {
    float m = -INFINITY;
    for (int l = 0; l < LTOT; ++l) m = fmaxf(m, sc[l]);
    float den = 0.f;
    for (int l = 0; l < LTOT; ++l) { float e = expf(sc[l] - m); sc[l] = e; den += e; }
    float inv = 1.f / den;
    for (int l = 0; l < LTOT; ++l) sc[l] *= inv;
  }
  __syncthreads();
  float c = 0.f;
  const float* ep = encf + (size_t)b * HIDD + tid;
  for (int l = 0; l < LTOT; ++l) c += sc[l] * ep[(size_t)l * BB * HIDD];
  u16 cb = f2bf(c);
  rin_c[(size_t)b * rinW + tid] = cb;
  pred_c[(size_t)b * predW + tid] = cb;
}
// GRU elementwise: h = (1-z)*n + z*hp
__global__ void gru_k(const float* __restrict__ gx, const float* __restrict__ gh,
                      const float* __restrict__ hp, float* __restrict__ hout,
                      u16* __restrict__ hbf, u16* __restrict__ predh, int predW) {
  int i = blockIdx.x * blockDim.x + threadIdx.x;
  if (i >= BB * HIDD) return;
  int b = i >> 7, c = i & 127;
  const float* gxb = gx + (size_t)b * 384;
  const float* ghb = gh + (size_t)b * 384;
  float r = 1.f / (1.f + expf(-(gxb[c]       + ghb[c])));
  float z = 1.f / (1.f + expf(-(gxb[c + 128] + ghb[c + 128])));
  float n = tanhf(gxb[c + 256] + r * ghb[c + 256]);
  float h = (1.f - z) * n + z * hp[i];
  hout[i] = h;
  if (hbf)   hbf[i] = f2bf(h);
  if (predh) predh[(size_t)b * predW + c] = f2bf(h);
}

// ---------------- host helpers ----------------
static inline void gemm(hipStream_t st, const u16* A, const u16* Bt, const float* bias,
                        float* C, int ldc, int M, int N, int K) {
  dim3 grid((N + 63) / 64, M / 128);
  size_t shmem = (size_t)64 * (K + 8) * sizeof(u16);
  gemm_wmma<<<grid, 128, shmem, st>>>(A, Bt, bias, C, ldc, M, N, K);
}
static inline void fillz(hipStream_t st, void* p, long long n32) {
  fill_u32_k<<<(unsigned)((n32 + 255) / 256), 256, 0, st>>>((u32*)p, 0u, n32);
}
static inline int cdiv(long long a, int b) { return (int)((a + b - 1) / b); }

extern "C" void kernel_launch(void* const* d_in, const int* in_sizes, int n_in,
                              void* d_out, int out_size, void* d_ws, size_t ws_size,
                              hipStream_t stream) {
  (void)in_sizes; (void)ws_size;
  const float* x0   = (const float*)d_in[0];
  const float* x1   = (const float*)d_in[1];
  const int*   ei0  = (const int*)d_in[2];
  const int*   ei1  = (const int*)d_in[3];
  const int*   Xact = (const int*)d_in[4];
  const int*   Xatr = (const int*)d_in[5];
  const unsigned char* mask = (const unsigned char*)d_in[6];
  // params: jax pytree order (dict keys sorted): dec0[0..14], dec1[15..30], enc0[31..38], enc1[39..46]
  const float* P[48];
  for (int i = 0; i < 48 && (8 + i) < n_in; ++i) P[i] = (const float*)d_in[8 + i];

  // ---- workspace carve ----
  char* wp = (char*)d_ws;
  size_t off = 0;
  auto ALC = [&](size_t bytes) -> char* {
    char* p = wp + off;
    off += (bytes + 255) & ~(size_t)255;
    return p;
  };
  // bf16 weights
  u16* w1t[2]  = { (u16*)ALC(512 * 64 * 2),  (u16*)ALC(512 * 32 * 2) };
  u16* w2t[2]  = { (u16*)ALC(128 * 512 * 2), (u16*)ALC(128 * 512 * 2) };
  u16* wih0[2] = { (u16*)ALC(384 * 256 * 2), (u16*)ALC(384 * 384 * 2) };
  u16* whh0[2] = { (u16*)ALC(384 * 128 * 2), (u16*)ALC(384 * 128 * 2) };
  u16* wih1[2] = { (u16*)ALC(384 * 128 * 2), (u16*)ALC(384 * 128 * 2) };
  u16* whh1[2] = { (u16*)ALC(384 * 128 * 2), (u16*)ALC(384 * 128 * 2) };
  u16* wk[2]   = { (u16*)ALC(128 * 128 * 2), (u16*)ALC(128 * 128 * 2) };
  u16* wq[2]   = { (u16*)ALC(128 * 128 * 2), (u16*)ALC(128 * 128 * 2) };
  u16* wout[2] = { (u16*)ALC(65 * 384 * 2),  (u16*)ALC(33 * 512 * 2) };
  // encoder buffers
  u16*   xbf  = (u16*)ALC((size_t)NN * 64 * 2);
  float* h1   = (float*)ALC((size_t)NN * 512 * 4);
  float* agg1 = (float*)ALC((size_t)NN * 512 * 4);
  u16*   h1e  = (u16*)ALC((size_t)NN * 512 * 2);
  float* h2   = (float*)ALC((size_t)NN * 128 * 4);
  float* agg2 = (float*)ALC((size_t)NN * 128 * 4);
  float* asb  = (float*)ALC((size_t)NN * 4 * 4);
  float* adb  = (float*)ALC((size_t)NN * 4 * 4);
  u32*   maxb = (u32*)ALC((size_t)NN * 4 * 4);
  float* denm = (float*)ALC((size_t)NN * 4 * 4);
  float* logb = (float*)ALC((size_t)NET * 4 * 4);
  float* ebuf = (float*)ALC((size_t)NET * 4 * 4);
  // encoding + decoder buffers
  float* encf  = (float*)ALC((size_t)LTOT * BB * HIDD * 4);
  u16*   encb  = (u16*)ALC((size_t)LTOT * BB * HIDD * 2);
  float* Kb[2] = { (float*)ALC((size_t)LTOT * BB * HIDD * 4),
                   (float*)ALC((size_t)LTOT * BB * HIDD * 4) };
  float* sbuf = (float*)ALC((size_t)BB * HIDD * 4);
  u16*   sbf  = (u16*)ALC((size_t)BB * HIDD * 2);
  float* qbuf = (float*)ALC((size_t)BB * HIDD * 4);
  u16*   rin  = (u16*)ALC((size_t)BB * 384 * 2);
  u16*   pred = (u16*)ALC((size_t)BB * 512 * 2);
  float* gx   = (float*)ALC((size_t)BB * 384 * 4);
  float* gh   = (float*)ALC((size_t)BB * 384 * 4);
  float* h0f  = (float*)ALC((size_t)BB * HIDD * 4);
  u16*   h0b  = (u16*)ALC((size_t)BB * HIDD * 2);

  // ---- zero outputs (row 0 of each decoder output must be zero) ----
  fillz(stream, d_out, (long long)out_size);

  // ---- weight conversion ----
  transpose_bf_k<<<cdiv(64 * 512, 256), 256, 0, stream>>>(P[31], w1t[0], 64, 512);
  transpose_bf_k<<<cdiv(32 * 512, 256), 256, 0, stream>>>(P[39], w1t[1], 32, 512);
  transpose_bf_k<<<cdiv(512 * 128, 256), 256, 0, stream>>>(P[32], w2t[0], 512, 128);
  transpose_bf_k<<<cdiv(512 * 128, 256), 256, 0, stream>>>(P[40], w2t[1], 512, 128);
  for (int d = 0; d < 2; ++d) {
    const float** Q = P + (d ? 15 : 0);
    int rinW = d ? 384 : 256, predW = d ? 512 : 384, V = d ? 33 : 65;
    convbf_k<<<cdiv(384 * rinW, 256), 256, 0, stream>>>(Q[2], wih0[d], 384 * rinW);
    convbf_k<<<cdiv(384 * 128, 256), 256, 0, stream>>>(Q[0], whh0[d], 384 * 128);
    convbf_k<<<cdiv(384 * 128, 256), 256, 0, stream>>>(Q[3], wih1[d], 384 * 128);
    convbf_k<<<cdiv(384 * 128, 256), 256, 0, stream>>>(Q[1], whh1[d], 384 * 128);
    convbf_k<<<cdiv(128 * 128, 256), 256, 0, stream>>>(Q[4], wk[d], 128 * 128);
    convbf_k<<<cdiv(128 * 128, 256), 256, 0, stream>>>(Q[6], wq[d], 128 * 128);
    convbf_k<<<cdiv(V * predW, 256), 256, 0, stream>>>(Q[5], wout[d], V * predW);
  }

  // ---- encoders (2 graphs) ----
  for (int g = 0; g < 2; ++g) {
    const float* x  = g ? x1 : x0;
    const int*   ei = g ? ei1 : ei0;
    int D = g ? 32 : 64;
    const float** EP = P + (g ? 39 : 31);  // W1,W2,ad1,ad2,as1,as2,b1,b2

    prep_k<<<cdiv((long long)NN * D, 256), 256, 0, stream>>>(x, xbf, D);
    // GAT layer 1 (H=4, C=128)
    gemm(stream, xbf, w1t[g], nullptr, h1, 512, NN, 512, D);
    attvec_k<<<cdiv(NN * 4, 256), 256, 0, stream>>>(h1, EP[4], asb, 4, 128);
    attvec_k<<<cdiv(NN * 4, 256), 256, 0, stream>>>(h1, EP[2], adb, 4, 128);
    fillz(stream, maxb, (long long)NN * 4);
    fillz(stream, denm, (long long)NN * 4);
    edge_logit_max_k<<<cdiv((long long)NET * 4, 256), 256, 0, stream>>>(ei, asb, adb, logb, maxb, 4);
    edge_exp_k<<<cdiv((long long)NET * 4, 256), 256, 0, stream>>>(ei, logb, maxb, ebuf, denm, 4);
    fillz(stream, agg1, (long long)NN * 512);
    edge_aggr_k<<<NET, 256, 0, stream>>>(ei, h1, ebuf, denm, agg1, 4, 128);
    bias_elu_bf_k<<<cdiv((long long)NN * 512, 256), 256, 0, stream>>>(agg1, EP[6], h1e, 512);
    // GAT layer 2 (H=1, C=128)
    gemm(stream, h1e, w2t[g], nullptr, h2, 128, NN, 128, 512);
    attvec_k<<<cdiv(NN, 256), 256, 0, stream>>>(h2, EP[5], asb, 1, 128);
    attvec_k<<<cdiv(NN, 256), 256, 0, stream>>>(h2, EP[3], adb, 1, 128);
    fillz(stream, maxb, NN);
    fillz(stream, denm, NN);
    edge_logit_max_k<<<cdiv(NET, 256), 256, 0, stream>>>(ei, asb, adb, logb, maxb, 1);
    edge_exp_k<<<cdiv(NET, 256), 256, 0, stream>>>(ei, logb, maxb, ebuf, denm, 1);
    fillz(stream, agg2, (long long)NN * 128);
    edge_aggr_k<<<NET, 128, 0, stream>>>(ei, h2, ebuf, denm, agg2, 1, 128);
    enc_scatter_k<<<cdiv((long long)NN * 128, 256), 256, 0, stream>>>(agg2, EP[7], encf, encb, g);
  }

  // ---- hoisted K projections: K = enc @ Wk^T + bk  (identical every step) ----
  gemm(stream, encb, wk[0], P[11], Kb[0], HIDD, LTOT * BB, HIDD, HIDD);
  gemm(stream, encb, wk[1], P[26], Kb[1], HIDD, LTOT * BB, HIDD, HIDD);

  // ---- decoders ----
  const int NBH = cdiv(BB * HIDD, 256);
  for (int d = 0; d < 2; ++d) {
    const float** Q = P + (d ? 15 : 0);
    int rinW = d ? 384 : 256, predW = d ? 512 : 384, V = d ? 33 : 65;
    float* outB = (float*)d_out + (d ? (size_t)BB * SEQL * 65 : 0);
    int ldout = SEQL * V;
    mean_k<<<NBH, 256, 0, stream>>>(encf, sbuf, sbf, d);
    for (int t = 0; t < SEQL - 1; ++t) {
      if (d == 0) {
        // rin = [emb | c], pred = [h1 | c | emb]
        emb_k<<<NBH, 256, 0, stream>>>(Xact + t, Q[14], rin, rinW, pred + 256, predW);
      } else {
        // rin = [ea | ev | c], pred = [h1 | c | ea | ev]
        emb_k<<<NBH, 256, 0, stream>>>(Xact + t + 1, Q[14], rin, rinW, pred + 256, predW);
        emb_k<<<NBH, 256, 0, stream>>>(Xatr + t, Q[15], rin + 128, rinW, pred + 384, predW);
      }
      gemm(stream, sbf, wq[d], Q[13], qbuf, HIDD, BB, HIDD, HIDD);
      attn_k<<<BB, 128, 0, stream>>>(qbuf, Kb[d], encf, mask,
                                     rin + (d ? 256 : 128), rinW, pred + 128, predW);
      gemm(stream, rin, wih0[d], Q[9], gx, 384, BB, 384, rinW);
      gemm(stream, sbf, whh0[d], Q[7], gh, 384, BB, 384, HIDD);
      gru_k<<<NBH, 256, 0, stream>>>(gx, gh, sbuf, h0f, h0b, nullptr, 0);
      gemm(stream, h0b, wih1[d], Q[10], gx, 384, BB, 384, HIDD);
      gemm(stream, sbf, whh1[d], Q[8], gh, 384, BB, 384, HIDD);  // reads old s before overwrite
      gru_k<<<NBH, 256, 0, stream>>>(gx, gh, sbuf, sbuf, sbf, pred, predW);
      gemm(stream, pred, wout[d], Q[12], outB + (size_t)(t + 1) * V, ldout, BB, V, predW);
    }
  }
}
